// MultiOmicGATModule_52613349376330
// MI455X (gfx1250) — matrix operand order repaired and checked by reading, not compiled, not run on hardware
//
#include <hip/hip_runtime.h>
#include <hip/hip_bf16.h>
#include <cmath>

// ---------------------------------------------------------------------------
// Types for CDNA5 WMMA (wave32): v_wmma_f32_16x16x32_bf16
// ---------------------------------------------------------------------------
typedef __bf16 bf16_t;
typedef __attribute__((ext_vector_type(16))) __bf16 v16bf;
typedef __attribute__((ext_vector_type(8)))  __bf16 v8bf;
typedef __attribute__((ext_vector_type(8)))  float  v8f;

#define HID   256
#define HEADS 8

// ---------------------------------------------------------------------------
// Wave32 reduction
// ---------------------------------------------------------------------------
__device__ __forceinline__ float waveRedSum(float v) {
  v += __shfl_xor(v, 16, 32);
  v += __shfl_xor(v, 8, 32);
  v += __shfl_xor(v, 4, 32);
  v += __shfl_xor(v, 2, 32);
  v += __shfl_xor(v, 1, 32);
  return v;
}

// ---------------------------------------------------------------------------
// WMMA fragment loaders, per CDNA5 ISA 7.12.2 layouts.
// A (16-bit, 16x32): lanes 0-15 hold M=lane, K = {k0..k0+7} U {k0+16..k0+23};
//                    lanes 16-31 hold M=lane-16, K = {k0+8..k0+15} U {k0+24..k0+31}.
// B (16-bit, 32x16): lanes 0-15 hold N=lane, K = {k0..k0+7} U {k0+16..k0+23};
//                    lanes 16-31 hold N=lane-16, K = {k0+8..k0+15} U {k0+24..k0+31}.
// ---------------------------------------------------------------------------
__device__ __forceinline__ v16bf load_a_frag(const bf16_t* __restrict__ row, int k0, int half) {
  v8bf lo = *(const v8bf*)(row + k0 + half * 8);
  v8bf hi = *(const v8bf*)(row + k0 + 16 + half * 8);
  return __builtin_shufflevector(lo, hi, 0,1,2,3,4,5,6,7,8,9,10,11,12,13,14,15);
}
__device__ __forceinline__ v16bf load_b_frag(const bf16_t* __restrict__ row, int k0, int half) {
  v8bf lo = *(const v8bf*)(row + k0 + half * 8);
  v8bf hi = *(const v8bf*)(row + k0 + 16 + half * 8);
  return __builtin_shufflevector(lo, hi, 0,1,2,3,4,5,6,7,8,9,10,11,12,13,14,15);
}

// ---------------------------------------------------------------------------
// C[MxN] = scale * (A[MxK] @ B) + bias[N],  with BT[N][K] = B[K][N] (row-major)
// Block: 256 threads = 8 waves (2x4), block tile 64(M) x 128(N),
// wave tile 32x32 = 2x2 WMMA 16x16x32 bf16 tiles, f32 accumulate.
// ---------------------------------------------------------------------------
__global__ __launch_bounds__(256)
void wmma_gemm_bf16_kernel(const bf16_t* __restrict__ A, const bf16_t* __restrict__ BT,
                           const float* __restrict__ bias, float* __restrict__ C,
                           int M, int N, int K, float scale) {
  const int lane = threadIdx.x & 31;
  const int w    = threadIdx.x >> 5;
  const int wm   = w >> 2;       // 0..1
  const int wn   = w & 3;        // 0..3
  const int mBase = blockIdx.x * 64 + wm * 32;
  const int nBase = blockIdx.y * 128 + wn * 32;
  const int half = lane >> 4;
  const int lr   = lane & 15;

  int ar0 = mBase + lr;      if (ar0 > M - 1) ar0 = M - 1;
  int ar1 = mBase + 16 + lr; if (ar1 > M - 1) ar1 = M - 1;
  int br0 = nBase + lr;      if (br0 > N - 1) br0 = N - 1;
  int br1 = nBase + 16 + lr; if (br1 > N - 1) br1 = N - 1;

  const bf16_t* pa0 = A  + (size_t)ar0 * K;
  const bf16_t* pa1 = A  + (size_t)ar1 * K;
  const bf16_t* pb0 = BT + (size_t)br0 * K;
  const bf16_t* pb1 = BT + (size_t)br1 * K;

  v8f c00 = {}, c01 = {}, c10 = {}, c11 = {};

  for (int k0 = 0; k0 < K; k0 += 32) {
    v16bf a0 = load_a_frag(pa0, k0, half);
    v16bf a1 = load_a_frag(pa1, k0, half);
    v16bf b0 = load_b_frag(pb0, k0, half);
    v16bf b1 = load_b_frag(pb1, k0, half);
    if (k0 + 32 < K) {   // gfx1250 global_prefetch_b8 of next K chunk
      __builtin_prefetch(pa0 + k0 + 32, 0, 1);
      __builtin_prefetch(pa1 + k0 + 32, 0, 1);
      __builtin_prefetch(pb0 + k0 + 32, 0, 1);
      __builtin_prefetch(pb1 + k0 + 32, 0, 1);
    }
    c00 = __builtin_amdgcn_wmma_f32_16x16x32_bf16(false, a0, false, b0, (short)0, c00, false, false);
    c01 = __builtin_amdgcn_wmma_f32_16x16x32_bf16(false, a0, false, b1, (short)0, c01, false, false);
    c10 = __builtin_amdgcn_wmma_f32_16x16x32_bf16(false, a1, false, b0, (short)0, c10, false, false);
    c11 = __builtin_amdgcn_wmma_f32_16x16x32_bf16(false, a1, false, b1, (short)0, c11, false, false);
  }

  // C/D layout: VGPR r, lanes 0-15 -> (M=r, N=lane); lanes 16-31 -> (M=8+r, N=lane-16)
  const int colA = nBase + lr;
  const int colB = nBase + 16 + lr;
  const int row0 = mBase + half * 8;        // sub-tile sm=0
  const int row1 = mBase + 16 + half * 8;   // sub-tile sm=1
  const float bA = bias ? bias[colA] : 0.0f;
  const float bB = bias ? bias[colB] : 0.0f;
#pragma unroll
  for (int r = 0; r < 8; ++r) {
    int r0 = row0 + r, r1 = row1 + r;
    if (r0 < M) {
      C[(size_t)r0 * N + colA] = c00[r] * scale + bA;
      C[(size_t)r0 * N + colB] = c01[r] * scale + bB;
    }
    if (r1 < M) {
      C[(size_t)r1 * N + colA] = c10[r] * scale + bA;
      C[(size_t)r1 * N + colB] = c11[r] * scale + bB;
    }
  }
}

// ---------------------------------------------------------------------------
// Conversions
// ---------------------------------------------------------------------------
__global__ void cvt_bf16_kernel(const float* __restrict__ in, bf16_t* __restrict__ out, int n) {
  int t = blockIdx.x * blockDim.x + threadIdx.x;
  if (t < n) out[t] = (bf16_t)in[t];
}
// out[c*ostride + r] = in[r*cols + c]
__global__ void cvt_bf16_tr_kernel(const float* __restrict__ in, bf16_t* __restrict__ out,
                                   int rows, int cols, int ostride) {
  int t = blockIdx.x * blockDim.x + threadIdx.x;
  if (t >= rows * cols) return;
  int r = t / cols, c = t - r * cols;
  out[(size_t)c * ostride + r] = (bf16_t)in[t];
}
// out[r*ostride + c] = in[r*cols + c]  (padding pre-zeroed)
__global__ void cvt_bf16_pad_kernel(const float* __restrict__ in, bf16_t* __restrict__ out,
                                    int rows, int cols, int ostride) {
  int t = blockIdx.x * blockDim.x + threadIdx.x;
  if (t >= rows * cols) return;
  int r = t / cols, c = t - r * cols;
  out[(size_t)r * ostride + c] = (bf16_t)in[t];
}

// ---------------------------------------------------------------------------
// GATv2 edge kernels. One wave per edge; lane = head-dim (32), loop 8 heads.
// ---------------------------------------------------------------------------
__device__ __forceinline__ unsigned encodeOrderedF32(float x) {
  unsigned u = __float_as_uint(x);
  return (u & 0x80000000u) ? ~u : (u | 0x80000000u);
}
__device__ __forceinline__ float decodeOrderedF32(unsigned u) {
  u = (u & 0x80000000u) ? (u & 0x7FFFFFFFu) : ~u;
  return __uint_as_float(u);
}

__global__ void gat_logits_kernel(const float* __restrict__ HL, const float* __restrict__ HR,
                                  const int* __restrict__ ei, const float* __restrict__ att,
                                  float* __restrict__ logit, unsigned* __restrict__ mbuf, int E) {
  int e = (blockIdx.x * blockDim.x + threadIdx.x) >> 5;
  int lane = threadIdx.x & 31;
  if (e >= E) return;
  int s = ei[e], d = ei[E + e];
  const float* hl = HL + (size_t)s * HID;
  const float* hr = HR + (size_t)d * HID;
#pragma unroll
  for (int h = 0; h < HEADS; ++h) {
    float v = hl[h * 32 + lane] + hr[h * 32 + lane];
    v = v > 0.0f ? v : 0.2f * v;           // leaky_relu(0.2)
    v *= att[h * 32 + lane];
    v = waveRedSum(v);
    if (lane == 0) {
      logit[(size_t)e * HEADS + h] = v;
      atomicMax(&mbuf[d * HEADS + h], encodeOrderedF32(v));
    }
  }
}

__global__ void gat_softmax_kernel(float* __restrict__ logit, const int* __restrict__ ei,
                                   const unsigned* __restrict__ mbuf, float* __restrict__ denom, int E) {
  int t = blockIdx.x * blockDim.x + threadIdx.x;
  if (t >= E * HEADS) return;
  int e = t >> 3, h = t & 7;
  int d = ei[E + e];
  float m = decodeOrderedF32(mbuf[d * HEADS + h]);
  if (!(fabsf(m) < 3.0e38f)) m = 0.0f;     // isfinite(m) ? m : 0
  float ex = expf(logit[t] - m);
  logit[t] = ex;                            // overwrite with exp
  atomicAdd(&denom[d * HEADS + h], ex);
}

__global__ void gat_scatter_kernel(const float* __restrict__ HL, const float* __restrict__ ex,
                                   const int* __restrict__ ei, const float* __restrict__ denom,
                                   float* __restrict__ acc, int E) {
  int e = (blockIdx.x * blockDim.x + threadIdx.x) >> 5;
  int lane = threadIdx.x & 31;
  if (e >= E) return;
  int s = ei[e], d = ei[E + e];
  const float* hl = HL + (size_t)s * HID;
  float* ac = acc + (size_t)d * HID;
#pragma unroll
  for (int h = 0; h < HEADS; ++h) {
    float alpha = ex[(size_t)e * HEADS + h] / (denom[d * HEADS + h] + 1e-16f);
    atomicAdd(&ac[h * 32 + lane], alpha * hl[h * 32 + lane]);
  }
}

__global__ void bias_add_kernel(float* __restrict__ acc, const float* __restrict__ bias, int n) {
  int t = blockIdx.x * blockDim.x + threadIdx.x;
  if (t < n) acc[t] += bias[t & (HID - 1)];
}

// ---------------------------------------------------------------------------
// x = LayerNorm(x + elu(acc)) * g + b   (in place, wave per row)
// ---------------------------------------------------------------------------
__global__ void ln_residual_kernel(float* __restrict__ x, const float* __restrict__ acc,
                                   const float* __restrict__ g, const float* __restrict__ b, int rows) {
  int w = (blockIdx.x * blockDim.x + threadIdx.x) >> 5;
  int lane = threadIdx.x & 31;
  if (w >= rows) return;
  float* xr = x + (size_t)w * HID;
  const float* ar = acc + (size_t)w * HID;
  float y[8];
  float s = 0.0f;
#pragma unroll
  for (int j = 0; j < 8; ++j) {
    int c = lane + j * 32;
    float a = ar[c];
    float eac = a > 0.0f ? a : (expf(a) - 1.0f);  // elu
    y[j] = xr[c] + eac;
    s += y[j];
  }
  s = waveRedSum(s);
  float mu = s * (1.0f / HID);
  float vs = 0.0f;
#pragma unroll
  for (int j = 0; j < 8; ++j) { float dd = y[j] - mu; vs += dd * dd; }
  vs = waveRedSum(vs) * (1.0f / HID);
  float inv = rsqrtf(vs + 1e-5f);
#pragma unroll
  for (int j = 0; j < 8; ++j) {
    int c = lane + j * 32;
    xr[c] = (y[j] - mu) * inv * g[c] + b[c];
  }
}

// out = LayerNorm(Z) * g + b
__global__ void ln_out_kernel(const float* __restrict__ Z, const float* __restrict__ g,
                              const float* __restrict__ b, float* __restrict__ out, int rows) {
  int w = (blockIdx.x * blockDim.x + threadIdx.x) >> 5;
  int lane = threadIdx.x & 31;
  if (w >= rows) return;
  const float* zr = Z + (size_t)w * HID;
  float y[8];
  float s = 0.0f;
#pragma unroll
  for (int j = 0; j < 8; ++j) { y[j] = zr[lane + j * 32]; s += y[j]; }
  s = waveRedSum(s);
  float mu = s * (1.0f / HID);
  float vs = 0.0f;
#pragma unroll
  for (int j = 0; j < 8; ++j) { float dd = y[j] - mu; vs += dd * dd; }
  vs = waveRedSum(vs) * (1.0f / HID);
  float inv = rsqrtf(vs + 1e-5f);
  float* orow = out + (size_t)w * HID;
#pragma unroll
  for (int j = 0; j < 8; ++j) {
    int c = lane + j * 32;
    orow[c] = (y[j] - mu) * inv * g[c] + b[c];
  }
}

// ---------------------------------------------------------------------------
// Host orchestration.
// Input flattening (JAX sorted-dict pytree order), 180 tensors:
//  0..2   batch_gene, batch_meth(cpg), batch_mirna
//  3..14  edge_indices, keys sorted:
//         cpg__coregulates__mirna, cpg__regulates__gene, cpg__self_loop__cpg,
//         gene__copathway__gene, gene__ppi__gene, gene__regulated_by__cpg,
//         gene__self_loop__gene, gene__targeted_by__mirna, mirna__coregulates__cpg,
//         mirna__samefamily__mirna, mirna__self_loop__mirna, mirna__targets__gene
//  15..86  conv layer0: per edge key {Wl, Wr, att, bias, bl, br}
//  87..158 conv layer1
//  159..161 emb: cpg, gene, mirna
//  162..167 ln layer0: cpg{b,g}, gene{b,g}, mirna{b,g};  168..173 ln layer1
//  174..179 out_ln: cpg{b,g}, gene{b,g}, mirna{b,g}
// node type ids: 0=gene, 1=cpg, 2=mirna
// ---------------------------------------------------------------------------
extern "C" void kernel_launch(void* const* d_in, const int* in_sizes, int n_in,
                              void* d_out, int out_size, void* d_ws, size_t ws_size,
                              hipStream_t stream) {
  (void)n_in; (void)out_size; (void)ws_size;
  static const int E_SRC[12]   = {1,1,1,0,0,0,0,0,2,2,2,2};
  static const int E_DST[12]   = {2,0,1,0,0,1,0,2,1,2,2,0};
  static const int EMB_IDX[3]  = {160, 159, 161};  // gene, cpg, mirna
  static const int BATCH_IDX[3]= {0, 1, 2};
  static const int LN_OFF[3]   = {2, 0, 4};        // gene, cpg, mirna within ln group

  int NT[3], KP[3];
  for (int nt = 0; nt < 3; ++nt) { NT[nt] = in_sizes[EMB_IDX[nt]] / HID; KP[nt] = (NT[nt] + 31) & ~31; }

  // ---- workspace bump allocator -------------------------------------------
  size_t off = 0;
  auto alloc = [&](size_t bytes) -> void* {
    void* p = (char*)d_ws + off;
    off += (bytes + 255) & ~(size_t)255;
    return p;
  };
  float*  x[3]; float* acc[3]; bf16_t* xbf[3]; bf16_t* xT[3]; bf16_t* bbf[3];
  for (int nt = 0; nt < 3; ++nt) x[nt]   = (float*)alloc((size_t)NT[nt] * HID * 4);
  for (int nt = 0; nt < 3; ++nt) acc[nt] = (float*)alloc((size_t)NT[nt] * HID * 4);
  for (int nt = 0; nt < 3; ++nt) xbf[nt] = (bf16_t*)alloc((size_t)NT[nt] * HID * 2);
  for (int nt = 0; nt < 3; ++nt) xT[nt]  = (bf16_t*)alloc((size_t)HID * KP[nt] * 2);
  for (int nt = 0; nt < 3; ++nt) bbf[nt] = (bf16_t*)alloc((size_t)64 * KP[nt] * 2);
  bf16_t* WlT = (bf16_t*)alloc((size_t)HID * HID * 2);
  bf16_t* WrT = (bf16_t*)alloc((size_t)HID * HID * 2);
  int maxN = NT[0]; for (int nt = 1; nt < 3; ++nt) if (NT[nt] > maxN) maxN = NT[nt];
  int maxE = 0; for (int e = 0; e < 12; ++e) { int E = in_sizes[3 + e] / 2; if (E > maxE) maxE = E; }
  float*    HL    = (float*)alloc((size_t)maxN * HID * 4);
  float*    HR    = (float*)alloc((size_t)maxN * HID * 4);
  float*    exb   = (float*)alloc((size_t)maxE * HEADS * 4);
  unsigned* mbuf  = (unsigned*)alloc((size_t)maxN * HEADS * 4);
  float*    denom = (float*)alloc((size_t)maxN * HEADS * 4);
  float*    Z     = (float*)alloc((size_t)64 * HID * 4);

  // ---- x = embeddings ------------------------------------------------------
  for (int nt = 0; nt < 3; ++nt)
    hipMemcpyAsync(x[nt], d_in[EMB_IDX[nt]], (size_t)NT[nt] * HID * 4,
                   hipMemcpyDeviceToDevice, stream);

  // ---- 2 GNN layers --------------------------------------------------------
  for (int l = 0; l < 2; ++l) {
    for (int nt = 0; nt < 3; ++nt)
      hipMemsetAsync(acc[nt], 0, (size_t)NT[nt] * HID * 4, stream);
    for (int nt = 0; nt < 3; ++nt) {
      int n = NT[nt] * HID;
      cvt_bf16_kernel<<<(n + 255) / 256, 256, 0, stream>>>(x[nt], xbf[nt], n);
    }
    for (int e = 0; e < 12; ++e) {
      int s = E_SRC[e], d = E_DST[e];
      int E = in_sizes[3 + e] / 2;
      const int* ei = (const int*)d_in[3 + e];
      int pb = 15 + l * 72 + e * 6;
      const float* Wl   = (const float*)d_in[pb + 0];
      const float* Wr   = (const float*)d_in[pb + 1];
      const float* att  = (const float*)d_in[pb + 2];
      const float* bias = (const float*)d_in[pb + 3];
      const float* bl   = (const float*)d_in[pb + 4];
      const float* br   = (const float*)d_in[pb + 5];

      cvt_bf16_tr_kernel<<<(HID * HID + 255) / 256, 256, 0, stream>>>(Wl, WlT, HID, HID, HID);
      cvt_bf16_tr_kernel<<<(HID * HID + 255) / 256, 256, 0, stream>>>(Wr, WrT, HID, HID, HID);
      {
        dim3 g((unsigned)((NT[s] + 63) / 64), (unsigned)((HID + 127) / 128));
        wmma_gemm_bf16_kernel<<<g, 256, 0, stream>>>(xbf[s], WlT, bl, HL, NT[s], HID, HID, 1.0f);
      }
      {
        dim3 g((unsigned)((NT[d] + 63) / 64), (unsigned)((HID + 127) / 128));
        wmma_gemm_bf16_kernel<<<g, 256, 0, stream>>>(xbf[d], WrT, br, HR, NT[d], HID, HID, 1.0f);
      }
      hipMemsetAsync(mbuf,  0, (size_t)NT[d] * HEADS * 4, stream);
      hipMemsetAsync(denom, 0, (size_t)NT[d] * HEADS * 4, stream);
      gat_logits_kernel <<<(E + 7) / 8,            256, 0, stream>>>(HL, HR, ei, att, exb, mbuf, E);
      gat_softmax_kernel<<<(E * HEADS + 255) / 256,256, 0, stream>>>(exb, ei, mbuf, denom, E);
      gat_scatter_kernel<<<(E + 7) / 8,            256, 0, stream>>>(HL, exb, ei, denom, acc[d], E);
      bias_add_kernel   <<<(NT[d] * HID + 255)/256,256, 0, stream>>>(acc[d], bias, NT[d] * HID);
    }
    for (int nt = 0; nt < 3; ++nt) {
      int lb = 162 + l * 6 + LN_OFF[nt];
      const float* lnb = (const float*)d_in[lb];
      const float* lng = (const float*)d_in[lb + 1];
      ln_residual_kernel<<<(NT[nt] + 7) / 8, 256, 0, stream>>>(x[nt], acc[nt], lng, lnb, NT[nt]);
    }
  }

  // ---- output projections (order: gene, cpg, mirna) -----------------------
  for (int nt = 0; nt < 3; ++nt) {
    hipMemsetAsync(xT[nt],  0, (size_t)HID * KP[nt] * 2, stream);
    hipMemsetAsync(bbf[nt], 0, (size_t)64 * KP[nt] * 2, stream);
    int n1 = NT[nt] * HID;
    cvt_bf16_tr_kernel<<<(n1 + 255) / 256, 256, 0, stream>>>(x[nt], xT[nt], NT[nt], HID, KP[nt]);
    int n2 = 64 * NT[nt];
    cvt_bf16_pad_kernel<<<(n2 + 255) / 256, 256, 0, stream>>>(
        (const float*)d_in[BATCH_IDX[nt]], bbf[nt], 64, NT[nt], KP[nt]);
    dim3 g(1u, (unsigned)((HID + 127) / 128));
    wmma_gemm_bf16_kernel<<<g, 256, 0, stream>>>(bbf[nt], xT[nt], nullptr, Z,
                                                 64, HID, KP[nt], 1.0f / sqrtf((float)NT[nt]));
    int ob = 174 + LN_OFF[nt];
    const float* ognb = (const float*)d_in[ob];
    const float* ogng = (const float*)d_in[ob + 1];
    ln_out_kernel<<<(64 + 7) / 8, 256, 0, stream>>>(Z, ogng, ognb,
                                                    (float*)d_out + (size_t)nt * 64 * HID, 64);
  }
}
